// LinearGatedHyper_42271068127772
// MI455X (gfx1250) — compile-verified, matrix operand order checked by reading, not run on hardware
//
#include <hip/hip_runtime.h>

typedef __attribute__((ext_vector_type(16))) __bf16 v16bf;
typedef __attribute__((ext_vector_type(8)))  float  v8f;

#define IN_F   1024
#define OUT_F  1024
#define CTX    128
#define NHYP   4
#define NBR    16
#define BATCH  2048
#define KCH    (IN_F / 32)   // 32 K-chunks of 32
#define BT     (BATCH / 16)  // 128 batch tiles
#define MT     2             // bTiles per wave
#define NT     4             // o's per wave

// round-to-nearest-even f32 -> bf16
__device__ __forceinline__ unsigned short f2bf(float f) {
  union { float f; unsigned u; } v; v.f = f;
  unsigned r = v.u + 0x7FFFu + ((v.u >> 16) & 1u);
  return (unsigned short)(r >> 16);
}

// WMMA 16-bit A/B fragment layout (ISA 7.12.2): slot(0..15) within lane + half h -> K (0..31)
__device__ __forceinline__ int slot_to_kin(int slot, int h) {
  int v = slot >> 1, p = slot & 1;
  return (v < 4) ? (2 * v + p + 8 * h) : (16 + 2 * (v - 4) + p + 8 * h);
}

// xl = clamp(x, +/- logit(0.99)) -> bf16, stored pre-swizzled as A fragments.
// flat layout: ((bt*KCH + kc)*32 + lane)*16 + slot
__global__ void prep_xl(const float* __restrict__ x, unsigned short* __restrict__ xlA) {
  int t = blockIdx.x * blockDim.x + threadIdx.x;
  if (t >= BATCH * IN_F) return;
  int slot = t & 15;
  int lane = (t >> 4) & 31;
  int kc   = (t >> 9) & (KCH - 1);
  int bt   = t >> 14;
  int m = lane & 15, h = lane >> 4;
  int kin = slot_to_kin(slot, h);
  int b = bt * 16 + m;
  int i = kc * 32 + kin;
  float vx = x[b * IN_F + i];
  vx = fminf(fmaxf(vx, -4.5951198501345898f), 4.5951198501345898f);
  xlA[t] = f2bf(vx);
}

// weights[o, n, i] -> bf16 B fragments (lane column n = branch, same K pattern).
// flat layout: ((o*KCH + kc)*32 + lane)*16 + slot
__global__ void prep_w(const float* __restrict__ w, unsigned short* __restrict__ wB) {
  int t = blockIdx.x * blockDim.x + threadIdx.x;
  if (t >= OUT_F * NBR * IN_F) return;
  int slot = t & 15;
  int lane = (t >> 4) & 31;
  int kc   = (t >> 9) & (KCH - 1);
  int o    = t >> 14;
  int n = lane & 15, h = lane >> 4;
  int kin = slot_to_kin(slot, h);
  int i = kc * 32 + kin;
  wB[t] = f2bf(w[(o * NBR + n) * IN_F + i]);
}

// Branch index: bits_n = (hyp_w[o,n,:] . context[b,:]) > hyp_b[o,n]; idx = sum bits*2^n
__global__ void hyp_idx(const float* __restrict__ ctx, const float* __restrict__ hw,
                        const float* __restrict__ hb, unsigned char* __restrict__ idxbuf) {
  __shared__ float c[CTX];
  int b = blockIdx.x;
  int tid = threadIdx.x;           // 128 threads
  c[tid] = ctx[b * CTX + tid];
  __syncthreads();
  const float4* c4 = (const float4*)c;
  for (int o = tid; o < OUT_F; o += 128) {
    unsigned sel = 0;
#pragma unroll
    for (int n = 0; n < NHYP; ++n) {
      const float4* w4 = (const float4*)(hw + (size_t)(o * NHYP + n) * CTX);
      float acc = 0.f;
#pragma unroll 8
      for (int q = 0; q < CTX / 4; ++q) {
        float4 a = w4[q]; float4 d = c4[q];
        acc += a.x * d.x + a.y * d.y + a.z * d.z + a.w * d.w;
      }
      if (acc > hb[o * NHYP + n]) sel |= (1u << n);
    }
    idxbuf[b * OUT_F + o] = (unsigned char)sel;
  }
}

// One wave = 2 bTiles x 4 o's (8 WMMA tiles): per K-chunk load 2 A + 4 B
// fragments and issue 8 independent v_wmma_f32_16x16x32_bf16 chains.
// Epilogue: bias add, per-row branch gather through LDS, clamp, store.
__global__ void gated_gemm(const unsigned short* __restrict__ xlA,
                           const unsigned short* __restrict__ wB,
                           const float* __restrict__ bias,
                           const unsigned char* __restrict__ idxbuf,
                           float* __restrict__ out) {
  __shared__ float tileC[8][16][17];
  int wave = threadIdx.x >> 5;
  int lane = threadIdx.x & 31;
  int g   = blockIdx.x * 8 + wave;       // 16384 groups total
  int og  = g & (OUT_F / NT - 1);        // 256 o-groups
  int btg = g >> 8;                      // 64 bt-groups
  int o0  = og * NT;
  int bt0 = btg * MT;

  const unsigned short* A0 = xlA + (size_t)(bt0 + 0) * KCH * 512 + lane * 16;
  const unsigned short* A1 = xlA + (size_t)(bt0 + 1) * KCH * 512 + lane * 16;
  const unsigned short* Bb = wB + (size_t)o0 * KCH * 512 + lane * 16;

  v8f acc[MT][NT];
#pragma unroll
  for (int mt = 0; mt < MT; ++mt)
#pragma unroll
    for (int j = 0; j < NT; ++j) acc[mt][j] = (v8f){};

#pragma unroll 2
  for (int kc = 0; kc < KCH; ++kc) {
    v16bf a0 = *(const v16bf*)(A0 + (size_t)kc * 512);
    v16bf a1 = *(const v16bf*)(A1 + (size_t)kc * 512);
#pragma unroll
    for (int j = 0; j < NT; ++j) {
      v16bf bb = *(const v16bf*)(Bb + ((size_t)j * KCH + kc) * 512);
      acc[0][j] = __builtin_amdgcn_wmma_f32_16x16x32_bf16(
          false, a0, false, bb, (short)0, acc[0][j], false, false);
      acc[1][j] = __builtin_amdgcn_wmma_f32_16x16x32_bf16(
          false, a1, false, bb, (short)0, acc[1][j], false, false);
    }
  }

  // C layout: lanes 0-15 -> M=r, N=lane; lanes 16-31 -> M=r+8, N=lane-16
  int n = lane & 15, h = lane >> 4;
#pragma unroll
  for (int mt = 0; mt < MT; ++mt) {
#pragma unroll
    for (int j = 0; j < NT; ++j) {
      int o = o0 + j;
      float bv = bias[o * NBR + n];
#pragma unroll
      for (int r = 0; r < 8; ++r) {
        tileC[wave][r + 8 * h][n] = acc[mt][j][r] + bv;
      }
      __syncthreads();   // order LDS reuse across tiles (uniform across block)
      if (lane < 16) {
        int b = (bt0 + mt) * 16 + lane;
        int sel = idxbuf[(size_t)b * OUT_F + o];
        float v = tileC[wave][lane][sel];
        v = fminf(fmaxf(v, -2.f), 2.f);
        out[(size_t)b * OUT_F + o] = v;
      }
      __syncthreads();
    }
  }
}

extern "C" void kernel_launch(void* const* d_in, const int* in_sizes, int n_in,
                              void* d_out, int out_size, void* d_ws, size_t ws_size,
                              hipStream_t stream) {
  const float* x    = (const float*)d_in[0];
  const float* ctx  = (const float*)d_in[1];
  const float* w    = (const float*)d_in[2];
  const float* bias = (const float*)d_in[3];
  const float* hw   = (const float*)d_in[4];
  const float* hb   = (const float*)d_in[5];
  float* out = (float*)d_out;

  char* ws = (char*)d_ws;
  unsigned short* xlA = (unsigned short*)ws;                                  // 4 MB
  unsigned short* wB  = (unsigned short*)(ws + (size_t)BATCH * IN_F * 2);     // 32 MB
  unsigned char*  idx = (unsigned char*)(ws + (size_t)BATCH * IN_F * 2
                                            + (size_t)OUT_F * NBR * IN_F * 2); // 2 MB

  prep_xl<<<(BATCH * IN_F) / 256, 256, 0, stream>>>(x, xlA);
  prep_w<<<(OUT_F * NBR * IN_F) / 256, 256, 0, stream>>>(w, wB);
  hyp_idx<<<BATCH, 128, 0, stream>>>(ctx, hw, hb, idx);
  gated_gemm<<<(BT * OUT_F) / (8 * MT * NT), 256, 0, stream>>>(xlA, wB, bias, idx, out);
}
